// FDRNN_10264971838089
// MI455X (gfx1250) — compile-verified, matrix operand order checked by reading, not run on hardware
//
#include <hip/hip_runtime.h>
#include <hip/hip_bf16.h>

typedef __attribute__((ext_vector_type(16))) _Float16 v16h;
typedef __attribute__((ext_vector_type(8)))  _Float16 v8h;
typedef __attribute__((ext_vector_type(8)))  float    v8f;

// int4 exactly as the builtin prototype spells it (GCC vector_size style)
typedef int fd_v4i __attribute__((vector_size(4 * sizeof(int))));
typedef __attribute__((address_space(1))) fd_v4i fd_v4i_g;   // global
typedef __attribute__((address_space(3))) fd_v4i fd_v4i_l;   // LDS

#define BS_TOK   32768     // B*S
#define NB       64        // batch
#define SS       512       // seq len
#define NI       50
#define NK       4
#define HR       512       // rnn hidden
#define LDH      528       // LDS row stride (halves) for h: 1056B, 32B aligned

#define BM 128
#define BN 64
#define BK 32
#define LDT 48             // LDS tile row stride (halves): 96B, 32B aligned

// ---- CDNA5 async copy global->LDS (ASYNCcnt-tracked), with safe fallback ----
#if defined(__has_builtin)
#  if __has_builtin(__builtin_amdgcn_global_load_async_to_lds_b128)
#    define FD_ASYNC 1
#  endif
#endif
#ifndef FD_ASYNC
#  define FD_ASYNC 0
#endif

__device__ __forceinline__ void fd_cp16(_Float16* dst_lds, const _Float16* src_glb) {
#if FD_ASYNC
    __builtin_amdgcn_global_load_async_to_lds_b128(
        (fd_v4i_g*)src_glb, (fd_v4i_l*)dst_lds, 0, 0);
#else
    *(v8h*)dst_lds = *(const v8h*)src_glb;
#endif
}

__device__ __forceinline__ void fd_wait_async() {
#if FD_ASYNC
#  if __has_builtin(__builtin_amdgcn_s_wait_asynccnt)
    __builtin_amdgcn_s_wait_asynccnt(0);
#  else
    asm volatile("s_wait_asynccnt 0x0" ::: "memory");
#  endif
#endif
}

#if defined(__has_builtin) && __has_builtin(__builtin_amdgcn_tanhf)
#  define FD_TANH(x) __builtin_amdgcn_tanhf(x)
#elif defined(__has_builtin) && __has_builtin(__builtin_amdgcn_tanh_f32)
#  define FD_TANH(x) __builtin_amdgcn_tanh_f32(x)
#else
#  define FD_TANH(x) tanhf(x)
#endif

#define FD_WMMA(a, b, c) \
    __builtin_amdgcn_wmma_f32_16x16x32_f16(false, (a), false, (b), (short)0, (c), false, false)

// ---------------- weight convert f32[N,K] -> f16[N,KP] (zero-padded) ----------------
__global__ void fdrnn_cvt_w(const float* __restrict__ src, _Float16* __restrict__ dst,
                            int N, int K, int KP) {
    int idx = blockIdx.x * blockDim.x + threadIdx.x;
    if (idx >= N * KP) return;
    int n = idx / KP, k = idx - n * KP;
    dst[idx] = (k < K) ? (_Float16)src[n * K + k] : (_Float16)0.f;
}

// ---------------- fuzzy layer: z0[tok][j] = exp(-(x-mu)^2/sig), f16, padded to 256 ----
__global__ void fdrnn_fuzzy(const float* __restrict__ x, const float* __restrict__ fp,
                            _Float16* __restrict__ z0) {
    int idx = blockIdx.x * blockDim.x + threadIdx.x;   // over BS_TOK*256
    int tok = idx >> 8;
    int j   = idx & 255;                               // j = k*50 + i for j<200
    float v = 0.f;
    if (j < NI * NK) {
        int i = j % NI;
        float mu = fp[2 * j + 0];
        float sg = fp[2 * j + 1];
        float d  = x[tok * NI + i] - mu;
        v = __expf(-(d * d) / sg);
    }
    z0[idx] = (_Float16)v;
}

// ---------------- WMMA GEMM: C[M,N] = A[M,KP] * W[N,KP]^T (+bias0 +bias1) -------------
// Block: 256 thr (8 waves). Block tile 128x64, double-buffered async LDS staging.
// Wave: 32(M) x 32(N) = 4 wmma tiles per K-step.
__device__ __forceinline__ void fd_stage(_Float16* as, _Float16* bs,
                                         const _Float16* __restrict__ A,
                                         const _Float16* __restrict__ W,
                                         int mBase, int nBase, int KP, int kb, int tid) {
    #pragma unroll
    for (int i = 0; i < 2; ++i) {                      // A: 128 rows x 32 halves
        int c = tid + i * 256;
        int row = c >> 2, col = (c & 3) * 8;
        fd_cp16(&as[row * LDT + col], &A[(mBase + row) * KP + kb + col]);
    }
    {                                                  // B: 64 rows x 32 halves
        int row = tid >> 2, col = (tid & 3) * 8;
        fd_cp16(&bs[row * LDT + col], &W[(nBase + row) * KP + kb + col]);
    }
}

__global__ __launch_bounds__(256) void fdrnn_gemm_wmma(
    const _Float16* __restrict__ A, const _Float16* __restrict__ W,
    const float* __restrict__ bias0, const float* __restrict__ bias1,
    _Float16* __restrict__ Ch, float* __restrict__ Cf,
    int M, int N, int KP) {
    __shared__ _Float16 As[2][BM * LDT];
    __shared__ _Float16 Bs[2][BN * LDT];
    const int tid  = threadIdx.x;
    const int wave = tid >> 5, lane = tid & 31;
    const int mBase = blockIdx.y * BM, nBase = blockIdx.x * BN;
    const int mo = (wave & 3) * 32;        // wave M offset: two 16-row subtiles
    const int no = (wave >> 2) * 32;       // wave N offset: two 16-col subtiles
    const int mrow = lane & 15;
    const int khalf = (lane >> 4) * 16;

    v8f acc00 = {}, acc01 = {}, acc10 = {}, acc11 = {};

    fd_stage(As[0], Bs[0], A, W, mBase, nBase, KP, 0, tid);
    fd_wait_async();
    __syncthreads();

    int buf = 0;
    for (int kb = 0; kb < KP; kb += BK) {
        int nb = buf ^ 1;
        if (kb + BK < KP)
            fd_stage(As[nb], Bs[nb], A, W, mBase, nBase, KP, kb + BK, tid);
        const _Float16* as = As[buf];
        const _Float16* bs = Bs[buf];
        v16h a0 = *(const v16h*)&as[(mo + mrow) * LDT + khalf];
        v16h a1 = *(const v16h*)&as[(mo + 16 + mrow) * LDT + khalf];
        v16h b0 = *(const v16h*)&bs[(no + mrow) * LDT + khalf];
        v16h b1 = *(const v16h*)&bs[(no + 16 + mrow) * LDT + khalf];
        acc00 = FD_WMMA(a0, b0, acc00);
        acc01 = FD_WMMA(a0, b1, acc01);
        acc10 = FD_WMMA(a1, b0, acc10);
        acc11 = FD_WMMA(a1, b1, acc11);
        fd_wait_async();        // next tile resident in LDS
        __syncthreads();        // all waves done reading current buffer
        buf = nb;
    }

    const int n0 = nBase + no + (lane & 15);
    const int n1 = n0 + 16;
    float bA0 = bias0 ? bias0[n0] : 0.f, bA1 = bias0 ? bias0[n1] : 0.f;
    if (bias1) { bA0 += bias1[n0]; bA1 += bias1[n1]; }
    #pragma unroll
    for (int ti = 0; ti < 2; ++ti) {
        const v8f& c0 = ti ? acc10 : acc00;
        const v8f& c1 = ti ? acc11 : acc01;
        #pragma unroll
        for (int r = 0; r < 8; ++r) {
            int m = mBase + mo + ti * 16 + (lane >> 4) * 8 + r;
            float v0 = c0[r] + bA0;
            float v1 = c1[r] + bA1;
            if (Ch) { Ch[m * N + n0] = (_Float16)v0; Ch[m * N + n1] = (_Float16)v1; }
            if (Cf) { Cf[m * N + n0] = v0;           Cf[m * N + n1] = v1; }
        }
    }
}

// ---------------- RNN scan: one block owns 16 batch rows, h kept in LDS ---------------
// Block: 1024 thr (32 waves); wave w computes 16(batch) x 16(hidden cols w*16..) tile.
__global__ __launch_bounds__(1024) void fdrnn_rnn_scan(
    const float* __restrict__ pre,      // [B,S,HR] f32 (x·W_ih^T + b_ih + b_hh)
    const _Float16* __restrict__ Whh,   // [HR,HR] f16
    _Float16* __restrict__ yh,          // f16 out [B,S,HR] or null
    float* __restrict__ yf) {           // f32 out [B,S,HR] or null
    __shared__ _Float16 h[16 * LDH];
    const int tid  = threadIdx.x;
    const int wave = tid >> 5, lane = tid & 31;
    const int bBase = blockIdx.x * 16;
    for (int i = tid; i < 16 * LDH; i += 1024) h[i] = (_Float16)0.f;
    const int no    = wave * 16;
    const int mrow  = lane & 15;
    const int khalf = (lane >> 4) * 16;
    const _Float16* wrow = &Whh[(no + mrow) * HR + khalf];
    // one-time warm of this wave's weight slice into cache (off the t-critical path)
    #pragma unroll
    for (int kb = 0; kb < HR; kb += 64)
        __builtin_prefetch((const void*)(wrow + kb), 0, 2);
    __syncthreads();
    for (int t = 0; t < SS; ++t) {
        v8f acc = {};
        #pragma unroll 4
        for (int kb = 0; kb < HR; kb += 32) {
            v16h a = *(const v16h*)&h[mrow * LDH + kb + khalf];
            v16h b = *(const v16h*)&wrow[kb];
            acc = FD_WMMA(a, b, acc);
        }
        __syncthreads();                 // all h reads for step t done
        const int mloc = (lane >> 4) * 8;
        const int n    = no + (lane & 15);
        #pragma unroll
        for (int r = 0; r < 8; ++r) {
            int b   = bBase + mloc + r;
            int idx = (b * SS + t) * HR + n;
            float v = FD_TANH(pre[idx] + acc[r]);
            h[(mloc + r) * LDH + n] = (_Float16)v;
            if (yh) yh[idx] = (_Float16)v;
            if (yf) yf[idx] = v;
        }
        __syncthreads();                 // h updated before step t+1 reads
    }
}

extern "C" void kernel_launch(void* const* d_in, const int* in_sizes, int n_in,
                              void* d_out, int out_size, void* d_ws, size_t ws_size,
                              hipStream_t stream) {
    const float* x    = (const float*)d_in[0];
    const float* fp   = (const float*)d_in[1];
    const float* fc0w = (const float*)d_in[2];  const float* fc0b = (const float*)d_in[3];
    const float* fc1w = (const float*)d_in[4];  const float* fc1b = (const float*)d_in[5];
    const float* fc2w = (const float*)d_in[6];  const float* fc2b = (const float*)d_in[7];
    const float* fc3w = (const float*)d_in[8];  const float* fc3b = (const float*)d_in[9];
    const float* wih0 = (const float*)d_in[10]; const float* whh0 = (const float*)d_in[11];
    const float* bih0 = (const float*)d_in[12]; const float* bhh0 = (const float*)d_in[13];
    const float* wih1 = (const float*)d_in[14]; const float* whh1 = (const float*)d_in[15];
    const float* bih1 = (const float*)d_in[16]; const float* bhh1 = (const float*)d_in[17];
    float* out = (float*)d_out;

    char* ws = (char*)d_ws;
    size_t off = 0;
    auto alloc = [&](size_t bytes) -> void* {
        void* p = ws + off;
        off += (bytes + 255) & ~(size_t)255;
        return p;
    };
    _Float16* W0   = (_Float16*)alloc((size_t)1024 * 256 * 2);   // fc0 padded K 200->256
    _Float16* W1   = (_Float16*)alloc((size_t)1024 * 1024 * 2);
    _Float16* W2   = (_Float16*)alloc((size_t)1024 * 1024 * 2);
    _Float16* W3   = (_Float16*)alloc((size_t)128  * 1024 * 2);
    _Float16* Wi0  = (_Float16*)alloc((size_t)512  * 128  * 2);
    _Float16* Wh0  = (_Float16*)alloc((size_t)512  * 512  * 2);
    _Float16* Wi1  = (_Float16*)alloc((size_t)512  * 512  * 2);
    _Float16* Wh1  = (_Float16*)alloc((size_t)512  * 512  * 2);
    _Float16* z0   = (_Float16*)alloc((size_t)BS_TOK * 256 * 2); // fuzzy out; reused as a3
    _Float16* bufA = (_Float16*)alloc((size_t)BS_TOK * 1024 * 2);
    _Float16* bufB = (_Float16*)alloc((size_t)BS_TOK * 1024 * 2); // reused as y0 f16
    float*    pre  = (float*)   alloc((size_t)BS_TOK * HR * 4);   // reused by both layers
    (void)ws_size; (void)in_sizes; (void)n_in; (void)out_size;

    // 1) weights -> f16 (fc0 padded 200->256)
    auto cvt = [&](const float* s, _Float16* d, int N, int K, int KP) {
        int tot = N * KP;
        fdrnn_cvt_w<<<(tot + 255) / 256, 256, 0, stream>>>(s, d, N, K, KP);
    };
    cvt(fc0w, W0, 1024, 200, 256);
    cvt(fc1w, W1, 1024, 1024, 1024);
    cvt(fc2w, W2, 1024, 1024, 1024);
    cvt(fc3w, W3, 128, 1024, 1024);
    cvt(wih0, Wi0, 512, 128, 128);
    cvt(whh0, Wh0, 512, 512, 512);
    cvt(wih1, Wi1, 512, 512, 512);
    cvt(whh1, Wh1, 512, 512, 512);

    // 2) fuzzy layer
    fdrnn_fuzzy<<<(BS_TOK * 256) / 256, 256, 0, stream>>>(x, fp, z0);

    // 3) FC stack (all WMMA, f16 in/out)
    auto gemm = [&](const _Float16* A, const _Float16* W, const float* b0, const float* b1,
                    _Float16* Ch, float* Cf, int M, int N, int KP) {
        dim3 grid(N / BN, M / BM);
        fdrnn_gemm_wmma<<<grid, 256, 0, stream>>>(A, W, b0, b1, Ch, Cf, M, N, KP);
    };
    gemm(z0,   W0, fc0b, nullptr, bufA, nullptr, BS_TOK, 1024, 256);
    gemm(bufA, W1, fc1b, nullptr, bufB, nullptr, BS_TOK, 1024, 1024);
    gemm(bufB, W2, fc2b, nullptr, bufA, nullptr, BS_TOK, 1024, 1024);
    _Float16* a3 = z0;                                  // z0 dead -> reuse for fc3 out
    gemm(bufA, W3, fc3b, nullptr, a3, nullptr, BS_TOK, 128, 1024);

    // 4) RNN layer 0: input projection (f32 pre) then LDS-resident scan -> y0 (f16)
    gemm(a3, Wi0, bih0, bhh0, nullptr, pre, BS_TOK, HR, 128);
    _Float16* y0 = bufB;                                // bufB dead -> reuse
    fdrnn_rnn_scan<<<NB / 16, 1024, 0, stream>>>(pre, Wh0, y0, nullptr);

    // 5) RNN layer 1: projection from y0, scan -> d_out (f32)
    gemm(y0, Wi1, bih1, bhh1, nullptr, pre, BS_TOK, HR, 512);
    fdrnn_rnn_scan<<<NB / 16, 1024, 0, stream>>>(pre, Wh1, nullptr, out);
}
